// SGDAT_73615739453496
// MI455X (gfx1250) — compile-verified, matrix operand order checked by reading, not run on hardware
//
#include <hip/hip_runtime.h>
#include <hip/hip_bf16.h>

#define B_ 8
#define N_ 32768

typedef __attribute__((ext_vector_type(16))) _Float16 v16h;
typedef __attribute__((ext_vector_type(8)))  float    v8f;

union U16H { v16h v; unsigned u[8]; _Float16 h[16]; };

// K index (even base) for 16-bit A/B WMMA fragment: VGPR j holds K pair
// {k,k+1}; lanes 0-15 vs 16-31 select the +8 half (ISA 7.12.2).
__device__ __forceinline__ int kidx_of(int j, int hi) {
    return ((j < 4) ? (2 * j) : (2 * j + 8)) + 8 * hi;
}

// ---------------------------------------------------------------- stats ----
__global__ void k_stats(const float* __restrict__ pts, float* __restrict__ stats) {
    const int b = blockIdx.x, tid = threadIdx.x;
    const float* base = pts + (size_t)b * N_ * 9;
    __shared__ float rx[256], ry[256], rz[256];
    float sx = 0.f, sy = 0.f, sz = 0.f;
    for (int p = tid; p < N_; p += 256) {
        const float* q = base + (size_t)p * 9;
        sx += q[0]; sy += q[1]; sz += q[2];
    }
    rx[tid] = sx; ry[tid] = sy; rz[tid] = sz; __syncthreads();
    for (int s = 128; s > 0; s >>= 1) {
        if (tid < s) { rx[tid] += rx[tid + s]; ry[tid] += ry[tid + s]; rz[tid] += rz[tid + s]; }
        __syncthreads();
    }
    __shared__ float c[3];
    if (tid == 0) { c[0] = rx[0] / N_; c[1] = ry[0] / N_; c[2] = rz[0] / N_; }
    __syncthreads();
    float mx = 0.f;
    for (int p = tid; p < N_; p += 256) {
        const float* q = base + (size_t)p * 9;
        float dx = q[0] - c[0], dy = q[1] - c[1], dz = q[2] - c[2];
        mx = fmaxf(mx, dx * dx + dy * dy + dz * dz);
    }
    rx[tid] = mx; __syncthreads();
    for (int s = 128; s > 0; s >>= 1) { if (tid < s) rx[tid] = fmaxf(rx[tid], rx[tid + s]); __syncthreads(); }
    if (tid == 0) {
        stats[b * 4 + 0] = c[0]; stats[b * 4 + 1] = c[1]; stats[b * 4 + 2] = c[2];
        stats[b * 4 + 3] = sqrtf(rx[0]) + 1e-6f;
    }
}

// ------------------------------------------------- xyz_n + feat0 (9->64) ----
__global__ void k_enc0(const float* __restrict__ pts, const float* __restrict__ stats,
                       const float* __restrict__ w, const float* __restrict__ g,
                       const float* __restrict__ bb,
                       float* __restrict__ xyzn, float* __restrict__ feat0) {
    __shared__ float sw[576], sg[64], sb[64];
    const int tid = threadIdx.x;
    for (int i = tid; i < 576; i += 256) sw[i] = w[i];
    const float rs = rsqrtf(1.f + 1e-5f);
    if (tid < 64) { sg[tid] = g[tid] * rs; sb[tid] = bb[tid]; }
    __syncthreads();
    const int gp = blockIdx.x * 256 + tid;
    const int b = gp >> 15;
    const float* q = pts + (size_t)gp * 9;
    float in[9];
#pragma unroll
    for (int i = 0; i < 9; ++i) in[i] = q[i];
    const float* st = stats + b * 4;
    const float inv = 1.f / st[3];
    xyzn[(size_t)gp * 3 + 0] = (in[0] - st[0]) * inv;
    xyzn[(size_t)gp * 3 + 1] = (in[1] - st[1]) * inv;
    xyzn[(size_t)gp * 3 + 2] = (in[2] - st[2]) * inv;
    float* fo = feat0 + (size_t)gp * 64;
    for (int c = 0; c < 64; ++c) {
        float a = 0.f;
#pragma unroll
        for (int k = 0; k < 9; ++k) a += in[k] * sw[c * 9 + k];
        fo[c] = fmaxf(a * sg[c] + sb[c], 0.f);
    }
}

// ---------------------------------------------------------------- FPS ------
__global__ void k_fps(const float* __restrict__ xyz, float* __restrict__ mind,
                      int* __restrict__ idxo, int n, int np) {
    const int b = blockIdx.x, tid = threadIdx.x;
    const float* X = xyz + (size_t)b * n * 3;
    float* M = mind + (size_t)b * n;
    int* O = idxo + (size_t)b * np;
    for (int p = tid; p < n; p += 256) M[p] = 1e10f;
    __shared__ float sv[256]; __shared__ int si[256];
    __shared__ float c[3]; __shared__ int cur;
    if (tid == 0) cur = 0;
    __syncthreads();
    for (int it = 0; it < np; ++it) {
        if (tid == 0) { O[it] = cur; c[0] = X[cur * 3]; c[1] = X[cur * 3 + 1]; c[2] = X[cur * 3 + 2]; }
        __syncthreads();
        float best = -1.f; int bi = 0;
        for (int p = tid; p < n; p += 256) {
            float dx = X[p * 3] - c[0], dy = X[p * 3 + 1] - c[1], dz = X[p * 3 + 2] - c[2];
            float d = dx * dx + dy * dy + dz * dz;
            float m = fminf(M[p], d); M[p] = m;
            if (m > best) { best = m; bi = p; }
        }
        sv[tid] = best; si[tid] = bi; __syncthreads();
        for (int s = 128; s > 0; s >>= 1) {
            if (tid < s && sv[tid + s] > sv[tid]) { sv[tid] = sv[tid + s]; si[tid] = si[tid + s]; }
            __syncthreads();
        }
        if (tid == 0) cur = si[0];
        __syncthreads();
    }
}

// ------------------------------------------------------------- gather3 -----
__global__ void k_gather3(const float* __restrict__ src, const int* __restrict__ idx,
                          float* __restrict__ dst, int src_n, int np, int total) {
    const int t = blockIdx.x * 256 + threadIdx.x;
    if (t >= total) return;
    const int b = t / np;
    const int s = idx[t];
    const float* p = src + ((size_t)b * src_n + s) * 3;
    float* q = dst + (size_t)t * 3;
    q[0] = p[0]; q[1] = p[1]; q[2] = p[2];
}

// ------------------------------------- level MLP: [xyz(3),feat(64)] -> C ---
__global__ void k_enc_level(const float* __restrict__ xyz_s, const int* __restrict__ idx,
                            const float* __restrict__ feat_src, int src_n,
                            const float* __restrict__ w, const float* __restrict__ g,
                            const float* __restrict__ bb, float* __restrict__ out, int np) {
    __shared__ float sin_[67];
    const int t = blockIdx.x, tid = threadIdx.x, C = blockDim.x;
    const int b = t / np;
    if (tid < 3) sin_[tid] = xyz_s[(size_t)t * 3 + tid];
    const int s = idx[t];
    const float* fs = feat_src + ((size_t)b * src_n + s) * 64;
    for (int k = tid; k < 64; k += C) sin_[3 + k] = fs[k];
    __syncthreads();
    const float* wr = w + tid * 67;
    float a = 0.f;
    for (int k = 0; k < 67; ++k) a += sin_[k] * wr[k];
    const float rs = rsqrtf(1.f + 1e-5f);
    out[(size_t)t * C + tid] = fmaxf(a * (g[tid] * rs) + bb[tid], 0.f);
}

// ------------------- 512-level fuse: nn-interp(128) + pos512 + up1 ---------
__global__ void k_fuse512(const float* __restrict__ xyz512, const float* __restrict__ xyz128,
                          const float* __restrict__ feat512, const float* __restrict__ feat128,
                          const float* __restrict__ posw, const float* __restrict__ posg,
                          const float* __restrict__ posb,
                          const float* __restrict__ up1w, const float* __restrict__ up1bias,
                          const float* __restrict__ up1g, const float* __restrict__ up1b,
                          float* __restrict__ fuse) {
    __shared__ float sin_[320];
    __shared__ float dv[128]; __shared__ int di[128];
    const int t = blockIdx.x, tid = threadIdx.x;
    const int b = t >> 9;
    const float x = xyz512[(size_t)t * 3], y = xyz512[(size_t)t * 3 + 1], z = xyz512[(size_t)t * 3 + 2];
    {
        const float* xk = xyz128 + ((size_t)(b * 128 + tid)) * 3;
        float dx = x - xk[0], dy = y - xk[1], dz = z - xk[2];
        dv[tid] = dx * dx + dy * dy + dz * dz; di[tid] = tid;
    }
    __syncthreads();
    for (int s = 64; s > 0; s >>= 1) {
        if (tid < s && dv[tid + s] < dv[tid]) { dv[tid] = dv[tid + s]; di[tid] = di[tid + s]; }
        __syncthreads();
    }
    const int nn = di[0];
    if (tid < 64) sin_[tid] = feat512[(size_t)t * 64 + tid];
    sin_[64 + tid] = feat128[((size_t)(b * 128 + nn)) * 128 + tid];
    const float rs = rsqrtf(1.f + 1e-5f);
    {
        const float* wr = posw + tid * 3;
        float a = x * wr[0] + y * wr[1] + z * wr[2];
        sin_[192 + tid] = fmaxf(a * (posg[tid] * rs) + posb[tid], 0.f);
    }
    __syncthreads();
    const float* wr = up1w + tid * 320;
    float a = up1bias[tid];
    for (int k = 0; k < 320; ++k) a += sin_[k] * wr[k];
    fuse[(size_t)t * 128 + tid] = fmaxf(a * (up1g[tid] * rs) + up1b[tid], 0.f);
}

// ------------------------------------------- nn index for all N points -----
__global__ void k_nnidx(const float* __restrict__ xyzn, const float* __restrict__ xyz512,
                        int* __restrict__ nni) {
    __shared__ float sx[512 * 3];
    const int bpb = N_ / 256;
    const int b = blockIdx.x / bpb, blk = blockIdx.x % bpb;
    for (int i = threadIdx.x; i < 512 * 3; i += 256) sx[i] = xyz512[(size_t)b * 512 * 3 + i];
    __syncthreads();
    const int gp = b * N_ + blk * 256 + threadIdx.x;
    const float x = xyzn[(size_t)gp * 3], y = xyzn[(size_t)gp * 3 + 1], z = xyzn[(size_t)gp * 3 + 2];
    float bd = 3.4e38f; int bi = 0;
    for (int j = 0; j < 512; ++j) {
        float dx = x - sx[j * 3], dy = y - sx[j * 3 + 1], dz = z - sx[j * 3 + 2];
        float d = dx * dx + dy * dy + dz * dz;
        if (d < bd) { bd = d; bi = j; }
    }
    nni[gp] = bi;
}

// ----------------- pack up2/head weights into WMMA B-fragment order --------
__global__ void k_pack(const float* __restrict__ up2w, const float* __restrict__ headw,
                       _Float16* __restrict__ pku, _Float16* __restrict__ pkh) {
    const int t = blockIdx.x * 256 + threadIdx.x;
    if (t < 32768) {
        int p = t & 1, j = (t >> 1) & 7, lane = (t >> 4) & 31, k = (t >> 9) & 7, tt = t >> 12;
        int n = lane & 15, hi = lane >> 4;
        int kk = k * 32 + kidx_of(j, hi) + p;
        pku[t] = (_Float16)up2w[(tt * 16 + n) * 256 + kk];
    } else if (t < 32768 + 2048) {
        int t2 = t - 32768;
        int p = t2 & 1, j = (t2 >> 1) & 7, lane = (t2 >> 4) & 31, k = (t2 >> 9) & 3;
        int n = lane & 15, hi = lane >> 4;
        int kk = k * 32 + kidx_of(j, hi) + p;
        pkh[t2] = (n < 13) ? (_Float16)headw[n * 128 + kk] : (_Float16)0.0f;
    }
}

// --------- fused tail: stage A -> WMMA up2(256->128) -> WMMA head(->13) ----
__global__ __launch_bounds__(128) void k_final(
    const float* __restrict__ xyzn, const float* __restrict__ feat0,
    const float* __restrict__ fuse512, const int* __restrict__ nnidx,
    const float* __restrict__ posNw, const float* __restrict__ posNg,
    const float* __restrict__ posNb,
    const _Float16* __restrict__ pku, const float* __restrict__ up2bias,
    const float* __restrict__ up2g, const float* __restrict__ up2b,
    const _Float16* __restrict__ pkh, const float* __restrict__ headb,
    float* __restrict__ out) {
    __shared__ _Float16 sA[4][16 * 256];                 // 32 KB, 4 waves/block
    const int tid = threadIdx.x, w = tid >> 5, lane = tid & 31;
    _Float16* A = &sA[w][0];
    const int gp0 = (blockIdx.x * 4 + w) * 16;           // 16-point tile
    const float rs = rsqrtf(1.0f + 1e-5f);
    const int hi = lane >> 4, nloc = lane & 15;

    // ---- stage A tile: K 0..127 = fuse512[nn], 128..191 = feat0, 192..255 = posN
    if (lane < 16) {
        const int m = lane, gp = gp0 + m;
        const int b = gp >> 15;
        const int nn = nnidx[gp];
        const float* fr = fuse512 + ((size_t)(b * 512 + nn)) * 128;
        _Float16* Ar = A + m * 256;
        for (int k = 0; k < 128; ++k) Ar[k] = (_Float16)fr[k];
        const float* f0 = feat0 + (size_t)gp * 64;
        for (int k = 0; k < 64; ++k) Ar[128 + k] = (_Float16)f0[k];
    } else {
        const int m = lane - 16, gp = gp0 + m;
        const float x = xyzn[(size_t)gp * 3], y = xyzn[(size_t)gp * 3 + 1], z = xyzn[(size_t)gp * 3 + 2];
        _Float16* Ar = A + m * 256;
        for (int c = 0; c < 64; ++c) {
            float a = x * posNw[c * 3] + y * posNw[c * 3 + 1] + z * posNw[c * 3 + 2];
            Ar[192 + c] = (_Float16)fmaxf(a * (posNg[c] * rs) + posNb[c], 0.0f);
        }
    }
    asm volatile("s_wait_dscnt 0" ::: "memory");         // same-wave LDS stage fence

    const unsigned* Au = (const unsigned*)A;
    v8f acc[8] = {};
#pragma unroll
    for (int ks = 0; ks < 8; ++ks) {
        U16H a;
#pragma unroll
        for (int j = 0; j < 8; ++j)
            a.u[j] = Au[(nloc * 256 + ks * 32 + kidx_of(j, hi)) >> 1];
#pragma unroll
        for (int t = 0; t < 8; ++t) {
            U16H bm;
            const uint4* bp = (const uint4*)(pku + (size_t)(((t * 8 + ks) * 32 + lane) * 16));
            uint4 b0 = bp[0], b1 = bp[1];
            bm.u[0] = b0.x; bm.u[1] = b0.y; bm.u[2] = b0.z; bm.u[3] = b0.w;
            bm.u[4] = b1.x; bm.u[5] = b1.y; bm.u[6] = b1.z; bm.u[7] = b1.w;
            acc[t] = __builtin_amdgcn_wmma_f32_16x16x32_f16(
                false, a.v, false, bm.v, (short)0, acc[t], false, false);
        }
    }
    // up2 bias + BN-ReLU, stash fuse_N (f16) back into LDS rows (cols 0..127)
#pragma unroll
    for (int t = 0; t < 8; ++t) {
        const int c = t * 16 + nloc;
        const float s = up2g[c] * rs, sh = up2b[c], bi = up2bias[c];
#pragma unroll
        for (int r = 0; r < 8; ++r) {
            const int m = r + 8 * hi;
            A[m * 256 + c] = (_Float16)fmaxf((acc[t][r] + bi) * s + sh, 0.0f);
        }
    }
    asm volatile("s_wait_dscnt 0" ::: "memory");

    v8f hacc = {};
#pragma unroll
    for (int ks = 0; ks < 4; ++ks) {
        U16H a;
#pragma unroll
        for (int j = 0; j < 8; ++j)
            a.u[j] = Au[(nloc * 256 + ks * 32 + kidx_of(j, hi)) >> 1];
        U16H bm;
        const uint4* bp = (const uint4*)(pkh + (size_t)((ks * 32 + lane) * 16));
        uint4 b0 = bp[0], b1 = bp[1];
        bm.u[0] = b0.x; bm.u[1] = b0.y; bm.u[2] = b0.z; bm.u[3] = b0.w;
        bm.u[4] = b1.x; bm.u[5] = b1.y; bm.u[6] = b1.z; bm.u[7] = b1.w;
        hacc = __builtin_amdgcn_wmma_f32_16x16x32_f16(
            false, a.v, false, bm.v, (short)0, hacc, false, false);
    }
    if (nloc < 13) {
        const float hb = headb[nloc];
#pragma unroll
        for (int r = 0; r < 8; ++r) {
            const int m = r + 8 * hi;
            out[(size_t)(gp0 + m) * 13 + nloc] = hacc[r] + hb;
        }
    }
}

// ===========================================================================
extern "C" void kernel_launch(void* const* d_in, const int* in_sizes, int n_in,
                              void* d_out, int out_size, void* d_ws, size_t ws_size,
                              hipStream_t stream) {
    (void)in_sizes; (void)n_in; (void)out_size; (void)ws_size;
    const float* pts      = (const float*)d_in[0];
    const float* enc0_w   = (const float*)d_in[1];
    const float* enc0_g   = (const float*)d_in[2];
    const float* enc0_b   = (const float*)d_in[3];
    const float* enc512_w = (const float*)d_in[4];
    const float* enc512_g = (const float*)d_in[5];
    const float* enc512_b = (const float*)d_in[6];
    const float* enc128_w = (const float*)d_in[7];
    const float* enc128_g = (const float*)d_in[8];
    const float* enc128_b = (const float*)d_in[9];
    const float* pos512_w = (const float*)d_in[10];
    const float* pos512_g = (const float*)d_in[11];
    const float* pos512_b = (const float*)d_in[12];
    const float* posN_w   = (const float*)d_in[13];
    const float* posN_g   = (const float*)d_in[14];
    const float* posN_b   = (const float*)d_in[15];
    const float* up1_w    = (const float*)d_in[16];
    const float* up1_bias = (const float*)d_in[17];
    const float* up1_g    = (const float*)d_in[18];
    const float* up1_b    = (const float*)d_in[19];
    const float* up2_w    = (const float*)d_in[20];
    const float* up2_bias = (const float*)d_in[21];
    const float* up2_g    = (const float*)d_in[22];
    const float* up2_b    = (const float*)d_in[23];
    const float* head_w   = (const float*)d_in[24];
    const float* head_b   = (const float*)d_in[25];

    char* ws = (char*)d_ws;
    size_t off = 0;
    auto alloc = [&](size_t bytes) -> void* {
        void* p = ws + off;
        off += (bytes + 255) & ~(size_t)255;
        return p;
    };
    float*     stats   = (float*)alloc((size_t)B_ * 4 * 4);
    float*     xyzn    = (float*)alloc((size_t)B_ * N_ * 3 * 4);
    float*     feat0   = (float*)alloc((size_t)B_ * N_ * 64 * 4);
    float*     mind    = (float*)alloc((size_t)B_ * N_ * 4);
    int*       idx512  = (int*)alloc((size_t)B_ * 512 * 4);
    float*     xyz512  = (float*)alloc((size_t)B_ * 512 * 3 * 4);
    float*     feat512 = (float*)alloc((size_t)B_ * 512 * 64 * 4);
    int*       idx128  = (int*)alloc((size_t)B_ * 128 * 4);
    float*     xyz128  = (float*)alloc((size_t)B_ * 128 * 3 * 4);
    float*     feat128 = (float*)alloc((size_t)B_ * 128 * 128 * 4);
    float*     fuse512 = (float*)alloc((size_t)B_ * 512 * 128 * 4);
    int*       nnidx   = (int*)alloc((size_t)B_ * N_ * 4);
    _Float16*  pku     = (_Float16*)alloc((size_t)32768 * 2);
    _Float16*  pkh     = (_Float16*)alloc((size_t)2048 * 2);

    k_stats<<<B_, 256, 0, stream>>>(pts, stats);
    k_enc0<<<(B_ * N_) / 256, 256, 0, stream>>>(pts, stats, enc0_w, enc0_g, enc0_b, xyzn, feat0);
    k_fps<<<B_, 256, 0, stream>>>(xyzn, mind, idx512, N_, 512);
    k_gather3<<<(B_ * 512 + 255) / 256, 256, 0, stream>>>(xyzn, idx512, xyz512, N_, 512, B_ * 512);
    k_enc_level<<<B_ * 512, 64, 0, stream>>>(xyz512, idx512, feat0, N_, enc512_w, enc512_g, enc512_b, feat512, 512);
    k_fps<<<B_, 256, 0, stream>>>(xyz512, mind, idx128, 512, 128);
    k_gather3<<<(B_ * 128 + 255) / 256, 256, 0, stream>>>(xyz512, idx128, xyz128, 512, 128, B_ * 128);
    k_enc_level<<<B_ * 128, 128, 0, stream>>>(xyz128, idx128, feat512, 512, enc128_w, enc128_g, enc128_b, feat128, 128);
    k_fuse512<<<B_ * 512, 128, 0, stream>>>(xyz512, xyz128, feat512, feat128,
                                            pos512_w, pos512_g, pos512_b,
                                            up1_w, up1_bias, up1_g, up1_b, fuse512);
    k_nnidx<<<B_ * (N_ / 256), 256, 0, stream>>>(xyzn, xyz512, nnidx);
    k_pack<<<(32768 + 2048 + 255) / 256, 256, 0, stream>>>(up2_w, head_w, pku, pkh);
    k_final<<<(B_ * N_) / 64, 128, 0, stream>>>(xyzn, feat0, fuse512, nnidx,
                                                posN_w, posN_g, posN_b,
                                                pku, up2_bias, up2_g, up2_b,
                                                pkh, head_b, (float*)d_out);
}
